// MultiheadAttention_66314295050284
// MI455X (gfx1250) — matrix operand order repaired and checked by reading, not compiled, not run on hardware
//
#include <hip/hip_runtime.h>
#include <math.h>

typedef __bf16 bf16;
typedef __attribute__((ext_vector_type(16))) __bf16 v16bf;
typedef __attribute__((ext_vector_type(8)))  __bf16 v8bf;
typedef __attribute__((ext_vector_type(8)))  float   v8f;
typedef int v4i __attribute__((vector_size(16)));

#define DEV __device__ __forceinline__

#if defined(__has_builtin)
#  if __has_builtin(__builtin_amdgcn_global_load_async_to_lds_b128)
#    define HAS_ASYNC_LDS 1
#  else
#    define HAS_ASYNC_LDS 0
#  endif
#else
#  define HAS_ASYNC_LDS 0
#endif

// ---------- helpers ----------

DEV bf16 f2bf(float f) {
  unsigned u = __builtin_bit_cast(unsigned, f);
  unsigned r = (u + 0x7FFFu + ((u >> 16) & 1u)) >> 16;   // round-to-nearest-even
  unsigned short h = (unsigned short)r;
  return __builtin_bit_cast(bf16, h);
}

DEV v8f wmma_bf16(v16bf a, v16bf b, v8f c) {
  // D(16x16 f32) = A(16x32 bf16) * B(32x16 bf16) + C
  return __builtin_amdgcn_wmma_f32_16x16x32_bf16(
      /*neg_a=*/false, a, /*neg_b=*/false, b,
      /*c_mod=*/(short)0, c, /*reuse_a=*/false, /*reuse_b=*/false);
}

// A-fragment (16x32, bf16), source row-major [M x K], base at (row0,k0).
// Lane L<16: row=L, K chunks [0..7],[16..23]; L>=16: row=L-16, [8..15],[24..31].
DEV v16bf load_a_frag(const bf16* __restrict__ base, int ld) {
  int lane = threadIdx.x & 31;
  const bf16* p = base + (lane & 15) * ld + ((lane >> 4) << 3);
  v16bf f;
  ((v8bf*)&f)[0] = *(const v8bf*)(p);
  ((v8bf*)&f)[1] = *(const v8bf*)(p + 16);
  return f;
}

// B-fragment (32x16, bf16) loaded from B^T stored row-major [N x K], base at (n0,k0).
// Lane L<16: col=L, K chunks [0..7],[16..23]; L>=16: col=L-16, [8..15],[24..31].
DEV v16bf load_bt_frag(const bf16* __restrict__ base, int ld) {
  int lane = threadIdx.x & 31;
  const bf16* p = base + (lane & 15) * ld + ((lane >> 4) << 3);
  v16bf f;
  ((v8bf*)&f)[0] = *(const v8bf*)(p);
  ((v8bf*)&f)[1] = *(const v8bf*)(p + 16);
  return f;
}

// B-fragment from an LDS tile laid out [64 rows][40 bf16] (padded vs 32 to
// spread the 64 LDS banks: 20-dword row stride -> 16 distinct banks).
DEV v16bf load_blds_frag(const bf16* base /* tile + j*16*40 */) {
  int lane = threadIdx.x & 31;
  const bf16* p = base + (lane & 15) * 40 + ((lane >> 4) << 3);
  v16bf f;
  ((v8bf*)&f)[0] = *(const v8bf*)(p);        // ds_load_b128
  ((v8bf*)&f)[1] = *(const v8bf*)(p + 16);
  return f;
}

// Stage a 64(N) x 32(K) bf16 tile of W (B^T, row-major, ld elements) into an
// LDS tile [64][40]. 128 threads x 2 chunks of 16B, coalesced (4 threads/row).
DEV void stage_b_tile(const bf16* __restrict__ gsrc, int ld, bf16* ltile) {
#pragma unroll
  for (int t = 0; t < 2; t++) {
    int q = threadIdx.x + t * 128;       // 0..255 chunk id
    int row = q >> 2, piece = q & 3;
    const bf16* g = gsrc + row * ld + piece * 8;
    bf16* l = ltile + row * 40 + piece * 8;
#if HAS_ASYNC_LDS
    __builtin_amdgcn_global_load_async_to_lds_b128(
        (__attribute__((address_space(1))) v4i*)(unsigned long long)g,
        (__attribute__((address_space(3))) v4i*)(unsigned int)(unsigned long long)l,
        0, 0);
#else
    *(v8bf*)l = *(const v8bf*)g;         // global b128 -> ds_store_b128
#endif
  }
}

DEV void wait_stage_and_barrier() {
#if HAS_ASYNC_LDS
#  if __has_builtin(__builtin_amdgcn_s_wait_asynccnt)
  __builtin_amdgcn_s_wait_asynccnt(0);
#  else
  asm volatile("s_wait_asynccnt 0x0" ::: "memory");
#  endif
#endif
  __syncthreads();
}

// Shared GEMM core: block = 128 threads (4 waves). Block tile 128(M) x 64(N),
// wave tile 32x64 (8 WMMAs / k-step). B tile double-buffered in LDS via
// async-to-LDS; A fragments direct global b128.
DEV void gemm_core_128x64(const bf16* __restrict__ xp /* X + row0*Kd */,
                          const bf16* __restrict__ wp /* W + n0*Kd (block) */,
                          int Kd, bf16* bt0, bf16* bt1, v8f acc[2][4]) {
  stage_b_tile(wp, Kd, bt0);
  for (int k0 = 0; k0 < Kd; k0 += 32) {
    bf16* cur = ((k0 >> 5) & 1) ? bt1 : bt0;
    bf16* nxt = ((k0 >> 5) & 1) ? bt0 : bt1;
    wait_stage_and_barrier();                    // cur ready; nxt free
    if (k0 + 32 < Kd) stage_b_tile(wp + k0 + 32, Kd, nxt);
    __builtin_prefetch(xp + k0 + 64, 0, 1);
    v16bf a0 = load_a_frag(xp + k0, Kd);
    v16bf a1 = load_a_frag(xp + 16 * Kd + k0, Kd);
#pragma unroll
    for (int j = 0; j < 4; j++) {
      v16bf bj = load_blds_frag(cur + j * 16 * 40);
      acc[0][j] = wmma_bf16(a0, bj, acc[0][j]);
      acc[1][j] = wmma_bf16(a1, bj, acc[1][j]);
    }
  }
}

// ---------- kernel 0: fp32 -> bf16 convert (x4 vectorized) ----------

__global__ __launch_bounds__(256) void cvt_f32_bf16_kernel(
    const float* __restrict__ src, bf16* __restrict__ dst, int n4) {
  int i = blockIdx.x * 256 + threadIdx.x;
  if (i < n4) {
    float4 v = ((const float4*)src)[i];
    bf16 r[4] = { f2bf(v.x), f2bf(v.y), f2bf(v.z), f2bf(v.w) };
    unsigned long long u;
    __builtin_memcpy(&u, r, 8);
    *(unsigned long long*)(dst + 4 * i) = u;
  }
}

// ---------- kernel 1: QKV projection GEMM ----------
// C[4096,3072] = X @ Wqkv^T + b; epilogue splits Q/K/V, folds 1/sqrt(64) into
// Q, writes Q,K as [B,H,S,64] bf16 and V transposed as [B,H,64,S] bf16.

__global__ __launch_bounds__(128) void qkv_gemm_kernel(
    const bf16* __restrict__ X, const bf16* __restrict__ W,
    const float* __restrict__ bias,
    bf16* __restrict__ Qb, bf16* __restrict__ Kb, bf16* __restrict__ Vt) {
  const int Kd = 1024, S = 2048, H = 16;
  __shared__ __align__(16) bf16 bt[2][64 * 40];

  int wave = threadIdx.x >> 5;
  int lane = threadIdx.x & 31;
  int hi   = lane >> 4;
  int c    = lane & 15;
  int row0 = blockIdx.x * 128 + wave * 32;
  int n0   = blockIdx.y * 64;

  v8f acc[2][4] = {};
  gemm_core_128x64(X + (long)row0 * Kd, W + (long)n0 * Kd, Kd,
                   bt[0], bt[1], acc);

  for (int j = 0; j < 4; j++) {
    int e = n0 + j * 16 + c;        // column in [0,3072)
    int h = e / 192;                // head
    int inner = e % 192;            // 0..63 q, 64..127 k, 128..191 v
    float bb = bias[e];
    for (int i = 0; i < 2; i++) {
      for (int v = 0; v < 8; v++) {
        int m = row0 + i * 16 + v + hi * 8;    // token index
        int b_ = m >> 11;                       // / 2048
        int s  = m & 2047;
        float val = acc[i][j][v] + bb;
        if (inner < 64) {
          Qb[(((b_ * H + h) * S) + s) * 64 + inner] = f2bf(val * 0.125f);
        } else if (inner < 128) {
          Kb[(((b_ * H + h) * S) + s) * 64 + (inner - 64)] = f2bf(val);
        } else {
          Vt[((b_ * H + h) * 64 + (inner - 128)) * (long)S + s] = f2bf(val);
        }
      }
    }
  }
}

// ---------- kernel 2: causal flash attention ----------
// grid (S/64, B*H), 128 threads. Each wave owns 16 query rows, streams
// 32 keys per step with online softmax; P restaged through per-wave LDS.

__global__ __launch_bounds__(128) void flash_attn_kernel(
    const bf16* __restrict__ Qb, const bf16* __restrict__ Kb,
    const bf16* __restrict__ Vt, bf16* __restrict__ Vout) {
  const int S = 2048, HD = 64;
  __shared__ __align__(16) bf16 plds[4][16][40];

  int wave = threadIdx.x >> 5;
  int lane = threadIdx.x & 31;
  int hi   = lane >> 4;
  int c    = lane & 15;
  int bh   = blockIdx.y;                    // b*16 + h
  int q0   = blockIdx.x * 64 + wave * 16;   // first query row for this wave

  const bf16* Q = Qb + (long)bh * S * HD;
  const bf16* K = Kb + (long)bh * S * HD;
  const bf16* V = Vt + (long)bh * HD * S;   // [64, 2048] row-major

  // Q fragments resident for the whole loop (16x64 = two 16x32 A-frags)
  v16bf qf0 = load_a_frag(Q + q0 * HD + 0, HD);
  v16bf qf1 = load_a_frag(Q + q0 * HD + 32, HD);

  v8f o[4] = {};
  float mrow[8], lrow[8];
  for (int v = 0; v < 8; v++) { mrow[v] = -__builtin_inff(); lrow[v] = 0.f; }

  for (int j0 = 0; j0 <= q0 + 15; j0 += 32) {
    // ---- scores: S(16x32) = Q(16x64) @ K^T ----
    v8f s0 = {}, s1 = {};
    {
      v16bf k00 = load_bt_frag(K + (j0 + 0) * HD + 0, HD);
      v16bf k01 = load_bt_frag(K + (j0 + 0) * HD + 32, HD);
      v16bf k10 = load_bt_frag(K + (j0 + 16) * HD + 0, HD);
      v16bf k11 = load_bt_frag(K + (j0 + 16) * HD + 32, HD);
      s0 = wmma_bf16(qf0, k00, s0);
      s0 = wmma_bf16(qf1, k01, s0);
      s1 = wmma_bf16(qf0, k10, s1);
      s1 = wmma_bf16(qf1, k11, s1);
    }
    // ---- causal mask ----
    for (int v = 0; v < 8; v++) {
      int r = q0 + v + hi * 8;
      if (j0 + c > r)      s0[v] = -__builtin_inff();
      if (j0 + 16 + c > r) s1[v] = -__builtin_inff();
    }
    // ---- online softmax (rows live across 16-lane halves) ----
    for (int v = 0; v < 8; v++) {
      float mx = fmaxf(s0[v], s1[v]);
      mx = fmaxf(mx, __shfl_xor(mx, 8, 32));
      mx = fmaxf(mx, __shfl_xor(mx, 4, 32));
      mx = fmaxf(mx, __shfl_xor(mx, 2, 32));
      mx = fmaxf(mx, __shfl_xor(mx, 1, 32));
      float mnew  = fmaxf(mrow[v], mx);
      float scale = __expf(mrow[v] - mnew);
      float p0 = __expf(s0[v] - mnew);
      float p1 = __expf(s1[v] - mnew);
      float rs = p0 + p1;
      rs += __shfl_xor(rs, 8, 32);
      rs += __shfl_xor(rs, 4, 32);
      rs += __shfl_xor(rs, 2, 32);
      rs += __shfl_xor(rs, 1, 32);
      lrow[v] = lrow[v] * scale + rs;
      mrow[v] = mnew;
      for (int f = 0; f < 4; f++) o[f][v] *= scale;
      int rl = v + hi * 8;
      plds[wave][rl][c]      = f2bf(p0);   // ds_store_b16
      plds[wave][rl][16 + c] = f2bf(p1);
    }
    // ---- reload P as A-fragment (16x32) from this wave's LDS slab ----
    v16bf pa;
    {
      const bf16* pp = &plds[wave][lane & 15][hi << 3];
      ((v8bf*)&pa)[0] = *(const v8bf*)(pp);        // ds_load_b128
      ((v8bf*)&pa)[1] = *(const v8bf*)(pp + 16);
    }
    // ---- O(16x64) += P(16x32) @ V(32x64), V pre-transposed ----
    for (int f = 0; f < 4; f++) {
      v16bf vb = load_bt_frag(V + (f * 16) * (long)S + j0, S);
      o[f] = wmma_bf16(pa, vb, o[f]);
    }
  }

  // ---- normalize, write values as [B, S, H*64] bf16 ----
  int b_ = bh >> 4, h = bh & 15;
  for (int v = 0; v < 8; v++) {
    float inv = 1.0f / lrow[v];
    int r = q0 + v + hi * 8;
    long rowoff = ((long)(b_ * S + r)) * 1024 + h * 64;
    for (int f = 0; f < 4; f++)
      Vout[rowoff + f * 16 + c] = f2bf(o[f][v] * inv);
  }
}

// ---------- kernel 3: output projection ----------
// out[4096,1024] = values @ Wout^T + b_out, fp32 result.

__global__ __launch_bounds__(128) void out_gemm_kernel(
    const bf16* __restrict__ Vl, const bf16* __restrict__ W,
    const float* __restrict__ bias, float* __restrict__ out) {
  const int Kd = 1024;
  __shared__ __align__(16) bf16 bt[2][64 * 40];

  int wave = threadIdx.x >> 5;
  int lane = threadIdx.x & 31;
  int hi   = lane >> 4;
  int c    = lane & 15;
  int row0 = blockIdx.x * 128 + wave * 32;
  int n0   = blockIdx.y * 64;

  v8f acc[2][4] = {};
  gemm_core_128x64(Vl + (long)row0 * Kd, W + (long)n0 * Kd, Kd,
                   bt[0], bt[1], acc);

  for (int j = 0; j < 4; j++) {
    int n = n0 + j * 16 + c;
    float bb = bias[n];
    for (int i = 0; i < 2; i++)
      for (int v = 0; v < 8; v++) {
        int m = row0 + i * 16 + v + hi * 8;
        out[(long)m * 1024 + n] = acc[i][j][v] + bb;
      }
  }
}

// ---------- host launcher ----------

extern "C" void kernel_launch(void* const* d_in, const int* in_sizes, int n_in,
                              void* d_out, int out_size, void* d_ws, size_t ws_size,
                              hipStream_t stream) {
  (void)in_sizes; (void)n_in; (void)out_size; (void)ws_size;

  const float* x     = (const float*)d_in[0];   // [2,2048,1024]
  const float* w_qkv = (const float*)d_in[1];   // [3072,1024]
  const float* b_qkv = (const float*)d_in[2];   // [3072]
  const float* w_out = (const float*)d_in[3];   // [1024,1024]
  const float* b_out = (const float*)d_in[4];   // [1024]
  float* out = (float*)d_out;                   // [2,2048,1024] fp32

  const int M = 4096;            // B*S
  char* ws = (char*)d_ws;
  size_t off = 0;
  bf16* xb   = (bf16*)(ws + off); off += (size_t)M * 1024 * 2;          // 8 MB
  bf16* wqb  = (bf16*)(ws + off); off += (size_t)3072 * 1024 * 2;       // 6 MB
  bf16* wob  = (bf16*)(ws + off); off += (size_t)1024 * 1024 * 2;       // 2 MB
  bf16* Qb   = (bf16*)(ws + off); off += (size_t)M * 1024 * 2;          // 8 MB
  bf16* Kb   = (bf16*)(ws + off); off += (size_t)M * 1024 * 2;          // 8 MB
  bf16* Vt   = (bf16*)(ws + off); off += (size_t)M * 1024 * 2;          // 8 MB
  bf16* Valb = (bf16*)(ws + off); off += (size_t)M * 1024 * 2;          // 8 MB

  // 0) fp32 -> bf16 conversions
  {
    int n4;
    n4 = (M * 1024) / 4;
    cvt_f32_bf16_kernel<<<(n4 + 255) / 256, 256, 0, stream>>>(x, xb, n4);
    n4 = (3072 * 1024) / 4;
    cvt_f32_bf16_kernel<<<(n4 + 255) / 256, 256, 0, stream>>>(w_qkv, wqb, n4);
    n4 = (1024 * 1024) / 4;
    cvt_f32_bf16_kernel<<<(n4 + 255) / 256, 256, 0, stream>>>(w_out, wob, n4);
  }

  // 1) QKV projection (M=4096, N=3072, K=1024), split into Q/K/V^T
  qkv_gemm_kernel<<<dim3(M / 128, 3072 / 64), 128, 0, stream>>>(
      xb, wqb, b_qkv, Qb, Kb, Vt);

  // 2) causal flash attention over 32 (b,h) pairs x 32 query tiles
  flash_attn_kernel<<<dim3(2048 / 64, 32), 128, 0, stream>>>(Qb, Kb, Vt, Valb);

  // 3) output projection (M=4096, N=1024, K=1024) -> fp32
  out_gemm_kernel<<<dim3(M / 128, 1024 / 64), 128, 0, stream>>>(
      Valb, wob, b_out, out);
}
